// SparseMultiHeadAttention_53687091200326
// MI455X (gfx1250) — matrix operand order repaired and checked by reading, not compiled, not run on hardware
//
#include <hip/hip_runtime.h>
#include <hip/hip_bf16.h>

// ---------------------------------------------------------------------------
// Problem constants (from reference): B=2, N=2048, C=1024, H=16, D=64
// ---------------------------------------------------------------------------
#define BB 2
#define NN 2048
#define CC 1024
#define HH 16
#define DD 64

typedef __attribute__((ext_vector_type(16))) _Float16 v16h;
typedef __attribute__((ext_vector_type(8)))  _Float16 v8h;
typedef __attribute__((ext_vector_type(8)))  float    v8f;

union FragH { v16h v; v8h h[2]; };

__device__ __forceinline__ v8f wmma_f16(v16h a, v16h b, v8f c) {
    // D = A(16x32 f16) * B(32x16 f16) + C(16x16 f32)
    return __builtin_amdgcn_wmma_f32_16x16x32_f16(
        /*neg_a=*/false, a, /*neg_b=*/false, b,
        /*c_mod=*/(short)0, c, /*reuse_a=*/false, /*reuse_b=*/false);
}

// ---------------------------------------------------------------------------
// DPP16 ROW_XMASK cross-lane reduction (stays within 16-lane halves, VALU pipe,
// no ds_bpermute traffic).  dpp_ctrl = 0x160 | xor_mask.
// ---------------------------------------------------------------------------
template <int CTRL>
__device__ __forceinline__ float dpp_movf(float x) {
    int r = __builtin_amdgcn_update_dpp(
        0, __builtin_bit_cast(int, x), CTRL, 0xf, 0xf, true);
    return __builtin_bit_cast(float, r);
}
__device__ __forceinline__ float half_max16(float v) {
    v = fmaxf(v, dpp_movf<0x161>(v));   // xor 1
    v = fmaxf(v, dpp_movf<0x162>(v));   // xor 2
    v = fmaxf(v, dpp_movf<0x164>(v));   // xor 4
    v = fmaxf(v, dpp_movf<0x168>(v));   // xor 8
    return v;
}
__device__ __forceinline__ float half_sum16(float v) {
    v += dpp_movf<0x161>(v);
    v += dpp_movf<0x162>(v);
    v += dpp_movf<0x164>(v);
    v += dpp_movf<0x168>(v);
    return v;
}

// A-fragment: 16x32 f16, row-major with row stride ldk.
// Lane L: row = L%16 ; elems e=0..15 -> k = 16*(L/16)+e  (contiguous 32 bytes)
__device__ __forceinline__ v16h load_a_frag(const _Float16* base, int ldk) {
    int lane = threadIdx.x & 31;
    const _Float16* p = base + (lane & 15) * ldk + ((lane >> 4) << 4);
    FragH f;
    f.h[0] = *(const v8h*)(p);
    f.h[1] = *(const v8h*)(p + 8);
    return f.v;
}

// B-fragment: 32x16 f16 (K x N).  Source stored N-major: row n, K contiguous, stride ldk.
// Lane L: col n = L%16 ; elems e=0..15 -> k = 16*(L/16)+e  (contiguous 32 bytes)
__device__ __forceinline__ v16h load_b_frag(const _Float16* base, int ldk) {
    int lane = threadIdx.x & 31;
    const _Float16* p = base + (lane & 15) * ldk + ((lane >> 4) << 4);
    FragH f;
    f.h[0] = *(const v8h*)(p);
    f.h[1] = *(const v8h*)(p + 8);
    return f.v;
}

// ---------------------------------------------------------------------------
// Elementwise fp32 -> f16
// ---------------------------------------------------------------------------
__global__ __launch_bounds__(256) void k_f32_to_f16(const float* __restrict__ in,
                                                    _Float16* __restrict__ out, int n) {
    int i = blockIdx.x * 256 + threadIdx.x;
    if (i < n) out[i] = (_Float16)in[i];
}

// Transpose K x Nn fp32 -> Nn x K f16 (weights become N-major, K contiguous)
__global__ __launch_bounds__(256) void k_transpose_f16(const float* __restrict__ in,
                                                       _Float16* __restrict__ out,
                                                       int K, int Nn) {
    int idx = blockIdx.x * 256 + threadIdx.x;
    if (idx >= K * Nn) return;
    int n = idx / K;
    int k = idx - n * K;
    out[idx] = (_Float16)in[(size_t)k * Nn + n];
}

// ---------------------------------------------------------------------------
// WMMA GEMM:  Out[M,Nn] = A[M,K] * BT[Nn,K]^T + bias[Nn]
// Block tile 256x128x32, 256 threads = 8 waves (4x2), wave tile 64x64
// (16 wmma per 16 ds_load_b128 -> best compute density).
// ---------------------------------------------------------------------------
template <bool OUT_F32>
__global__ __launch_bounds__(256) void k_gemm_wmma(const _Float16* __restrict__ A,
                                                   const _Float16* __restrict__ BT,
                                                   const float* __restrict__ bias,
                                                   void* __restrict__ Out,
                                                   int M, int Nn, int K) {
    __shared__ _Float16 As[256 * 32];   // 16 KB
    __shared__ _Float16 Bs[128 * 32];   //  8 KB

    const int t    = threadIdx.x;
    const int lane = t & 31;
    const int wave = t >> 5;
    const int wm   = (wave >> 1) * 64;   // wave row offset in block tile
    const int wn   = (wave & 1) * 64;    // wave col offset in block tile
    const int m0   = blockIdx.y * 256;
    const int n0   = blockIdx.x * 128;

    v8f acc[4][4];
    for (int i = 0; i < 4; ++i)
        for (int j = 0; j < 4; ++j) acc[i][j] = {};

    for (int kt = 0; kt < K; kt += 32) {
        __syncthreads();
        // Stage A: 256x32 (1024 x 16B chunks), B: 128x32 (512 chunks).
        for (int c = t; c < 1024; c += 256) {
            int row = c >> 2;
            int kc  = (c & 3) << 3;
            const _Float16* gp = &A[(size_t)(m0 + row) * K + kt + kc];
            *(v8h*)&As[row * 32 + kc] = *(const v8h*)gp;
            if (kt + 32 < K) __builtin_prefetch(gp + 32, 0, 1);
        }
        for (int c = t; c < 512; c += 256) {
            int row = c >> 2;
            int kc  = (c & 3) << 3;
            const _Float16* gp = &BT[(size_t)(n0 + row) * K + kt + kc];
            *(v8h*)&Bs[row * 32 + kc] = *(const v8h*)gp;
            if (kt + 32 < K) __builtin_prefetch(gp + 32, 0, 1);
        }
        __syncthreads();

        v16h a0 = load_a_frag(As + (wm +  0) * 32, 32);
        v16h a1 = load_a_frag(As + (wm + 16) * 32, 32);
        v16h a2 = load_a_frag(As + (wm + 32) * 32, 32);
        v16h a3 = load_a_frag(As + (wm + 48) * 32, 32);
#pragma unroll
        for (int j = 0; j < 4; ++j) {
            v16h b = load_b_frag(Bs + (wn + j * 16) * 32, 32);
            acc[0][j] = wmma_f16(a0, b, acc[0][j]);
            acc[1][j] = wmma_f16(a1, b, acc[1][j]);
            acc[2][j] = wmma_f16(a2, b, acc[2][j]);
            acc[3][j] = wmma_f16(a3, b, acc[3][j]);
        }
    }

    // Epilogue: bias add + store (C-frag: vgpr r, lanes<16 -> M=r, lanes>=16 -> M=r+8)
#pragma unroll
    for (int i = 0; i < 4; ++i) {
#pragma unroll
        for (int j = 0; j < 4; ++j) {
            int col  = n0 + wn + j * 16 + (lane & 15);
            int rowb = m0 + wm + i * 16 + ((lane >> 4) << 3);
            float bv = bias[col];
#pragma unroll
            for (int r = 0; r < 8; ++r) {
                float v = acc[i][j][r] + bv;
                if (OUT_F32)
                    ((float*)Out)[(size_t)(rowb + r) * Nn + col] = v;
                else
                    ((_Float16*)Out)[(size_t)(rowb + r) * Nn + col] = (_Float16)v;
            }
        }
    }
}

// ---------------------------------------------------------------------------
// QK RMS-norm + scatter:  qkv[BN,3C] (f16, bias applied) ->
//   q [BH,N,D] = (q/rms)*gamma_q            (1/sqrt(D) attention scale folded)
//   k [BH,N,D] = (k/rms)*gamma_k*sqrt(D)
//   v [BH,D,N] (transposed for PV B-fragments)
// ---------------------------------------------------------------------------
__global__ __launch_bounds__(256) void k_qk_norm_scatter(const _Float16* __restrict__ qkv,
                                                         const float* __restrict__ gq,
                                                         const float* __restrict__ gk,
                                                         _Float16* __restrict__ qo,
                                                         _Float16* __restrict__ ko,
                                                         _Float16* __restrict__ vo) {
    int bn = blockIdx.x;
    int b  = bn / NN;
    int n  = bn - b * NN;
    int t  = threadIdx.x;
    int h  = t >> 4;
    int d0 = (t & 15) << 2;

    const _Float16* row = qkv + (size_t)bn * (3 * CC);
    float qv[4], kv[4], vv[4];
    float sq = 0.f, sk = 0.f;
#pragma unroll
    for (int i = 0; i < 4; ++i) {
        qv[i] = (float)row[h * DD + d0 + i];
        kv[i] = (float)row[CC + h * DD + d0 + i];
        vv[i] = (float)row[2 * CC + h * DD + d0 + i];
        sq += qv[i] * qv[i];
        sk += kv[i] * kv[i];
    }
    sq = half_sum16(sq);       // reduce over the 16 lanes of this head (DPP)
    sk = half_sum16(sk);
    float invq = rsqrtf(fmaxf(sq, 1e-24f));
    float invk = rsqrtf(fmaxf(sk, 1e-24f)) * 8.0f;  // * sqrt(D)

    size_t bh = (size_t)b * HH + h;
    _Float16* qp = qo + (bh * NN + n) * DD + d0;
    _Float16* kp = ko + (bh * NN + n) * DD + d0;
#pragma unroll
    for (int i = 0; i < 4; ++i) {
        qp[i] = (_Float16)(qv[i] * invq * gq[h * DD + d0 + i]);
        kp[i] = (_Float16)(kv[i] * invk * gk[h * DD + d0 + i]);
        vo[(bh * DD + d0 + i) * NN + n] = (_Float16)vv[i];
    }
}

// ---------------------------------------------------------------------------
// Flash attention (scale pre-folded into q).  One block = 128 q rows of one
// (b,h); 8 waves, each wave owns 16 q rows.  64-key tiles staged in LDS:
// 16 wmma per tile, softmax stats amortized over 64 keys, DPP reductions.
// ---------------------------------------------------------------------------
__global__ __launch_bounds__(256) void k_attention(const _Float16* __restrict__ Q,
                                                   const _Float16* __restrict__ Kmat,
                                                   const _Float16* __restrict__ Vt,
                                                   _Float16* __restrict__ Outp) {
    __shared__ _Float16 Ks[64 * 64];       // [key][d]      8 KB
    __shared__ _Float16 Vs[64 * 64];       // [d][key]      8 KB (Vt is [BH,D,N])
    __shared__ _Float16 Ps[8 * 16 * 64];   // per-wave P    16 KB, [row][key]

    const int t    = threadIdx.x;
    const int lane = t & 31;
    const int wave = t >> 5;
    const int bh   = blockIdx.y;
    const int b    = bh >> 4;
    const int h    = bh & 15;
    const int q0   = blockIdx.x * 128 + wave * 16;

    // Q fragments for d-steps 0,1 (A-frag layout, loaded straight from global)
    FragH qa[2];
    {
        const _Float16* qbase = Q + ((size_t)bh * NN + q0 + (lane & 15)) * DD;
#pragma unroll
        for (int s = 0; s < 2; ++s) {
            int koff = ((lane >> 4) << 4) + s * 32;
            qa[s].h[0] = *(const v8h*)(qbase + koff);
            qa[s].h[1] = *(const v8h*)(qbase + koff + 8);
        }
    }

    v8f acc[4];
    for (int j = 0; j < 4; ++j) acc[j] = {};
    float m_r[8], l_r[8];
#pragma unroll
    for (int r = 0; r < 8; ++r) { m_r[r] = -1e30f; l_r[r] = 0.f; }

    _Float16* myP = Ps + wave * (16 * 64);

    for (int kt = 0; kt < NN; kt += 64) {
        __syncthreads();
        {   // stage K (64x64) and V' (64x64): 512 chunks each, 2 per thread
            for (int c = t; c < 512; c += 256) {
                int kr = c >> 3, kc = (c & 7) << 3;
                *(v8h*)&Ks[kr * 64 + kc] =
                    *(const v8h*)&Kmat[((size_t)bh * NN + kt + kr) * DD + kc];
                int vd = c >> 3, vc = (c & 7) << 3;
                *(v8h*)&Vs[vd * 64 + vc] =
                    *(const v8h*)&Vt[((size_t)bh * DD + vd) * NN + kt + vc];
            }
        }
        __syncthreads();

        // S = Q * K^T : four 16-key column fragments
        v8f sc[4];
#pragma unroll
        for (int j = 0; j < 4; ++j) sc[j] = {};
#pragma unroll
        for (int s = 0; s < 2; ++s) {
#pragma unroll
            for (int j = 0; j < 4; ++j) {
                const _Float16* kb =
                    Ks + (j * 16 + (lane & 15)) * 64 + s * 32 + ((lane >> 4) << 4);
                FragH bf;
                bf.h[0] = *(const v8h*)(kb);
                bf.h[1] = *(const v8h*)(kb + 8);
                sc[j] = wmma_f16(qa[s].v, bf.v, sc[j]);
            }
        }

        // Online softmax per row r (row = r + 8*(lane/16); stats replicated/half)
#pragma unroll
        for (int r = 0; r < 8; ++r) {
            float mr = fmaxf(fmaxf(sc[0][r], sc[1][r]), fmaxf(sc[2][r], sc[3][r]));
            mr = half_max16(mr);
            float mnew = fmaxf(m_r[r], mr);
            float corr = __expf(m_r[r] - mnew);
            float p0 = __expf(sc[0][r] - mnew);
            float p1 = __expf(sc[1][r] - mnew);
            float p2 = __expf(sc[2][r] - mnew);
            float p3 = __expf(sc[3][r] - mnew);
            float rs = half_sum16((p0 + p1) + (p2 + p3));
            l_r[r] = l_r[r] * corr + rs;
            m_r[r] = mnew;
#pragma unroll
            for (int j = 0; j < 4; ++j) acc[j][r] *= corr;
            int row = r + ((lane >> 4) << 3);
            myP[row * 64 +      (lane & 15)] = (_Float16)p0;
            myP[row * 64 + 16 + (lane & 15)] = (_Float16)p1;
            myP[row * 64 + 32 + (lane & 15)] = (_Float16)p2;
            myP[row * 64 + 48 + (lane & 15)] = (_Float16)p3;
        }
        // Wave-private LDS RAW: DS ops are in-order within a wave -> no barrier.

        // acc += P(16x64) * V(64x64)
#pragma unroll
        for (int s = 0; s < 2; ++s) {
            v16h pa = load_a_frag(myP + s * 32, 64);
#pragma unroll
            for (int jd = 0; jd < 4; ++jd) {
                const _Float16* vb =
                    Vs + (jd * 16 + (lane & 15)) * 64 + s * 32 + ((lane >> 4) << 4);
                FragH vf;
                vf.h[0] = *(const v8h*)(vb);
                vf.h[1] = *(const v8h*)(vb + 8);
                acc[jd] = wmma_f16(pa, vf.v, acc[jd]);
            }
        }
    }

    // Write out: [B,N,C] f16, c = h*64 + jd*16 + lane%16
#pragma unroll
    for (int jd = 0; jd < 4; ++jd) {
        int col = h * DD + jd * 16 + (lane & 15);
#pragma unroll
        for (int r = 0; r < 8; ++r) {
            int row = q0 + r + ((lane >> 4) << 3);
            float v = acc[jd][r] / l_r[r];
            Outp[((size_t)b * NN + row) * CC + col] = (_Float16)v;
        }
    }
}

// ---------------------------------------------------------------------------
// Host launcher
// ---------------------------------------------------------------------------
extern "C" void kernel_launch(void* const* d_in, const int* in_sizes, int n_in,
                              void* d_out, int out_size, void* d_ws, size_t ws_size,
                              hipStream_t stream) {
    (void)in_sizes; (void)n_in; (void)out_size; (void)ws_size;

    const float* x     = (const float*)d_in[0];   // [B,N,C]
    const float* Wqkv  = (const float*)d_in[1];   // [C,3C]
    const float* bqkv  = (const float*)d_in[2];   // [3C]
    const float* gq    = (const float*)d_in[3];   // [H,D]
    const float* gk    = (const float*)d_in[4];   // [H,D]
    const float* Wout  = (const float*)d_in[5];   // [C,C]
    const float* bout  = (const float*)d_in[6];   // [C]
    float* out = (float*)d_out;                   // [B,N,C]

    // Workspace carve-up (all f16)
    char* w = (char*)d_ws;
    size_t off = 0;
    auto carve = [&](size_t elems) { _Float16* p = (_Float16*)(w + off); off += elems * 2; return p; };
    _Float16* x_h    = carve((size_t)BB * NN * CC);        // 8 MB
    _Float16* WqkvT  = carve((size_t)3 * CC * CC);         // 6 MB
    _Float16* WoutT  = carve((size_t)CC * CC);             // 2 MB
    _Float16* qkv_h  = carve((size_t)BB * NN * 3 * CC);    // 24 MB
    _Float16* q_h    = carve((size_t)BB * HH * NN * DD);   // 8 MB
    _Float16* k_h    = carve((size_t)BB * HH * NN * DD);   // 8 MB
    _Float16* v_h    = carve((size_t)BB * HH * DD * NN);   // 8 MB
    _Float16* ao_h   = carve((size_t)BB * NN * CC);        // 8 MB

    const int M = BB * NN;   // 4096

    // 1) precision conversion + weight transposes
    k_f32_to_f16<<<(M * CC + 255) / 256, 256, 0, stream>>>(x, x_h, M * CC);
    k_transpose_f16<<<(3 * CC * CC + 255) / 256, 256, 0, stream>>>(Wqkv, WqkvT, CC, 3 * CC);
    k_transpose_f16<<<(CC * CC + 255) / 256, 256, 0, stream>>>(Wout, WoutT, CC, CC);

    // 2) QKV projection: [4096,1024] x [1024,3072] + bias -> f16
    {
        dim3 grid(3 * CC / 128, M / 256);
        k_gemm_wmma<false><<<grid, 256, 0, stream>>>(x_h, WqkvT, bqkv, qkv_h, M, 3 * CC, CC);
    }

    // 3) RMS-norm q/k (scale folded) + scatter to attention layouts
    k_qk_norm_scatter<<<BB * NN, 256, 0, stream>>>(qkv_h, gq, gk, q_h, k_h, v_h);

    // 4) flash attention per (b,h)
    {
        dim3 grid(NN / 128, BB * HH);
        k_attention<<<grid, 256, 0, stream>>>(q_h, k_h, v_h, ao_h);
    }

    // 5) output projection: [4096,1024] x [1024,1024] + bias -> f32 d_out
    {
        dim3 grid(CC / 128, M / 256);
        k_gemm_wmma<true><<<grid, 256, 0, stream>>>(ao_h, WoutT, bout, out, M, CC, CC);
    }
}
